// EfficientAttention_15719580303663
// MI455X (gfx1250) — matrix operand order repaired and checked by reading, not compiled, not run on hardware
//
#include <hip/hip_runtime.h>

// ---------------------------------------------------------------------------
// GQA attention block for MI455X (gfx1250, wave32, WMMA).
// All GEMM work goes through v_wmma_f32_16x16x32_f16 (f16 in, f32 accum).
// 32(M)x64(N) tiles per wave: each B tile is reused by two A row-groups,
// cutting L2->register traffic to 1.5 b128 loads per WMMA.
// ---------------------------------------------------------------------------

#define S_LEN 2048
#define HIDN  2048
#define NH    32
#define NKV   8
#define DH    64

typedef __attribute__((ext_vector_type(16))) _Float16 v16h;
typedef __attribute__((ext_vector_type(8)))  _Float16 v8h;
typedef __attribute__((ext_vector_type(8)))  float    v8f;

// D = A(16x32 f16) * B(32x16 f16) + C(16x16 f32)
__device__ __forceinline__ v8f wmma_f16(v16h a, v16h b, v8f c) {
    return __builtin_amdgcn_wmma_f32_16x16x32_f16(
        /*neg_a=*/false, a, /*neg_b=*/false, b,
        /*c_mod=*/(short)0, c, /*reuse_a=*/false, /*reuse_b=*/false);
}

// A-operand loader: 16(M) x 32(K), row-major source (stride ld elements).
// Contiguous-k layout used here (not AMD's table order): lane L holds row M = L%16;
//   lanes 0-15:  halves 0-15 -> K=0..15
//   lanes 16-31: halves 0-15 -> K=16..31
__device__ __forceinline__ v16h load_a(const _Float16* base, int ld, int lane) {
    const int r = lane & 15, hf = lane >> 4;
    const _Float16* p = base + (size_t)r * ld;
    v8h lo = *(const v8h*)(p + hf * 16);
    v8h hi = *(const v8h*)(p + hf * 16 + 8);
    v16h out;
#pragma unroll
    for (int i = 0; i < 8; ++i) { out[i] = lo[i]; out[i + 8] = hi[i]; }
    return out;
}

// B-operand loader: 32(K) x 16(N), source stored as N rows x K-contiguous
// (stride ld elements). B layout used here (same k order as load_a): lane L holds column N = L%16,
// halves 0-15 -> K = (L/16)*16 + 0..15  (one 32B contiguous chunk).
__device__ __forceinline__ v16h load_b(const _Float16* base, int ld, int lane) {
    const int n = lane & 15, hf = lane >> 4;
    return *(const v16h*)(base + (size_t)n * ld + hf * 16);
}

// ---------------------------------------------------------------------------
__global__ void f32_to_f16_kernel(const float* __restrict__ in,
                                  _Float16* __restrict__ out, int n) {
    int i = blockIdx.x * blockDim.x + threadIdx.x;
    if (i < n) out[i] = (_Float16)in[i];
}

// ---------------------------------------------------------------------------
// QKV projection (+ optional RoPE). One wave computes a 32(s) x 64(d) tile
// for one head: two 16-row A groups share the four B (weight) tiles.
// mode 0: RoPE, store [h][s][d]. mode 1: no RoPE, store [h][d][s].
__global__ __launch_bounds__(32)
void proj_kernel(const _Float16* __restrict__ X,   // [S][HID] f16
                 const _Float16* __restrict__ W,   // [nh*DH][HID] f16
                 const float* __restrict__ cosb,   // [S][DH] (mode 0)
                 const float* __restrict__ sinb,   // [S][DH] (mode 0)
                 _Float16* __restrict__ out,
                 int mode) {
    const int lane = threadIdx.x;
    const int r = lane & 15, hf = lane >> 4;
    const int s0 = blockIdx.x * 32;
    const int h  = blockIdx.y;

    v8f acc[2][4] = {};
    const _Float16* xb = X + (size_t)s0 * HIDN;
    const _Float16* wb = W + (size_t)h * DH * HIDN;

    for (int k = 0; k < HIDN; k += 32) {
        v16h a0 = load_a(xb + k,                      HIDN, lane);  // rows s0..s0+15
        v16h a1 = load_a(xb + (size_t)16 * HIDN + k,  HIDN, lane);  // rows s0+16..s0+31
#pragma unroll
        for (int n = 0; n < 4; ++n) {
            v16h b = load_b(wb + (size_t)n * 16 * HIDN + k, HIDN, lane);
            acc[0][n] = wmma_f16(a0, b, acc[0][n]);
            acc[1][n] = wmma_f16(a1, b, acc[1][n]);
        }
    }

#pragma unroll
    for (int g = 0; g < 2; ++g) {
        if (mode == 0) {  // RoPE + row-major [h][s][d]
#pragma unroll
            for (int e = 0; e < 8; ++e) {
                const int srow = s0 + g * 16 + e + 8 * hf;
                const float* cr = cosb + (size_t)srow * DH;
                const float* sr = sinb + (size_t)srow * DH;
                float c0 = cr[r],      c1 = cr[16 + r], c2 = cr[32 + r], c3 = cr[48 + r];
                float n0 = sr[r],      n1 = sr[16 + r], n2 = sr[32 + r], n3 = sr[48 + r];
                float v0 = acc[g][0][e], v1 = acc[g][1][e];
                float v2 = acc[g][2][e], v3 = acc[g][3][e];
                // rotate_half: d<32 -> -x[d+32], d>=32 -> x[d-32]
                float o0 = v0 * c0 - v2 * n0;
                float o1 = v1 * c1 - v3 * n1;
                float o2 = v2 * c2 + v0 * n2;
                float o3 = v3 * c3 + v1 * n3;
                _Float16* ob = out + ((size_t)h * S_LEN + srow) * DH;
                ob[r]      = (_Float16)o0;
                ob[16 + r] = (_Float16)o1;
                ob[32 + r] = (_Float16)o2;
                ob[48 + r] = (_Float16)o3;
            }
        } else {          // V: no RoPE, transposed store [h][d][s]
#pragma unroll
            for (int e = 0; e < 8; ++e) {
                const int srow = s0 + g * 16 + e + 8 * hf;
#pragma unroll
                for (int dt = 0; dt < 4; ++dt) {
                    const int d = dt * 16 + r;
                    out[((size_t)h * DH + d) * S_LEN + srow] = (_Float16)acc[g][dt][e];
                }
            }
        }
    }
}

// ---------------------------------------------------------------------------
// Flash-style causal GQA attention. One wave: 32 query rows of one head
// (two 16-row groups sharing every K and V tile load).
__global__ __launch_bounds__(32)
void attn_kernel(const _Float16* __restrict__ Qh,  // [NH][S][DH]
                 const _Float16* __restrict__ Kh,  // [NKV][S][DH]
                 const _Float16* __restrict__ Vt,  // [NKV][DH][S]
                 _Float16* __restrict__ Ctx) {     // [S][NH*DH]
    __shared__ _Float16 pl[32 * 32];  // P tile staging (D-layout -> A-layout)

    const int lane = threadIdx.x;
    const int r = lane & 15, hf = lane >> 4;
    const int s0  = blockIdx.x * 32;
    const int h   = blockIdx.y;
    const int hkv = h >> 2;  // G = 4 query heads per kv head

    v16h aq[2][2];
#pragma unroll
    for (int g = 0; g < 2; ++g) {
        const _Float16* qbase = Qh + ((size_t)h * S_LEN + s0 + g * 16) * DH;
        aq[g][0] = load_a(qbase,      DH, lane);  // d 0..31
        aq[g][1] = load_a(qbase + 32, DH, lane);  // d 32..63
    }

    v8f acc[2][4] = {};
    float m[2][8], l[2][8];
#pragma unroll
    for (int g = 0; g < 2; ++g)
#pragma unroll
        for (int e = 0; e < 8; ++e) { m[g][e] = -1.0e30f; l[g][e] = 0.0f; }

    const int jmax = (s0 + 31) >> 5;  // inclusive; 32-wide key chunks
    for (int j = 0; j <= jmax; ++j) {
        const int t0 = j * 32;

        // --- scores: 32 x 32, K = 64 over D; K-tiles shared by both groups ---
        v8f c[2][2];
#pragma unroll
        for (int n = 0; n < 2; ++n) {
            const _Float16* kb = Kh + ((size_t)hkv * S_LEN + t0 + n * 16) * DH;
            v16h b0 = load_b(kb,      DH, lane);  // d 0..31
            v16h b1 = load_b(kb + 32, DH, lane);  // d 32..63
#pragma unroll
            for (int g = 0; g < 2; ++g) {
                v8f cc = {};
                cc = wmma_f16(aq[g][0], b0, cc);
                cc = wmma_f16(aq[g][1], b1, cc);
                c[g][n] = cc;
            }
        }

        // --- scale + causal mask (col = t0 + n*16 + r, row = s0 + g*16 + e + 8*hf) ---
#pragma unroll
        for (int g = 0; g < 2; ++g)
#pragma unroll
            for (int n = 0; n < 2; ++n)
#pragma unroll
                for (int e = 0; e < 8; ++e) {
                    const int t    = t0 + n * 16 + r;
                    const int srow = s0 + g * 16 + e + 8 * hf;
                    float v = c[g][n][e] * 0.125f;  // 1/sqrt(64)
                    c[g][n][e] = (t <= srow) ? v : -1.0e30f;
                }

        // --- online softmax + P into LDS ---
        __syncthreads();
#pragma unroll
        for (int g = 0; g < 2; ++g) {
#pragma unroll
            for (int e = 0; e < 8; ++e) {
                float mx = fmaxf(c[g][0][e], c[g][1][e]);
#pragma unroll
                for (int o = 1; o < 16; o <<= 1) mx = fmaxf(mx, __shfl_xor(mx, o, 32));
                float mn = fmaxf(m[g][e], mx);
                float sc = __expf(m[g][e] - mn);
                float p0 = __expf(c[g][0][e] - mn);
                float p1 = __expf(c[g][1][e] - mn);
                float rs = p0 + p1;
#pragma unroll
                for (int o = 1; o < 16; o <<= 1) rs += __shfl_xor(rs, o, 32);
                l[g][e] = l[g][e] * sc + rs;
                m[g][e] = mn;
#pragma unroll
                for (int dt = 0; dt < 4; ++dt) acc[g][dt][e] *= sc;
                const int row = g * 16 + e + 8 * hf;
                pl[row * 32 + r]      = (_Float16)p0;
                pl[row * 32 + 16 + r] = (_Float16)p1;
            }
        }
        __syncthreads();
        v16h pa0 = load_a(pl,           32, lane);
        v16h pa1 = load_a(pl + 16 * 32, 32, lane);

        // --- P (32x32) x V^T tiles (32x16); V-tiles shared by both groups ---
#pragma unroll
        for (int dt = 0; dt < 4; ++dt) {
            const _Float16* vb = Vt + ((size_t)hkv * DH + dt * 16) * S_LEN + t0;
            v16h b = load_b(vb, S_LEN, lane);
            acc[0][dt] = wmma_f16(pa0, b, acc[0][dt]);
            acc[1][dt] = wmma_f16(pa1, b, acc[1][dt]);
        }
        __syncthreads();
    }

    // --- normalize and emit ctx as [s][h*64 + d] f16 ---
#pragma unroll
    for (int g = 0; g < 2; ++g)
#pragma unroll
        for (int e = 0; e < 8; ++e) {
            const float inv = 1.0f / l[g][e];
            const int srow = s0 + g * 16 + e + 8 * hf;
#pragma unroll
            for (int dt = 0; dt < 4; ++dt) {
                const int d = dt * 16 + r;
                Ctx[(size_t)srow * (NH * DH) + h * DH + d] =
                    (_Float16)(acc[g][dt][e] * inv);
            }
        }
}

// ---------------------------------------------------------------------------
// Output projection: out[s, o] = sum_k ctx[s,k] * Wo[o,k]; f32 stores.
// 32(M)x64(N) per wave, B tiles shared across two A row groups.
__global__ __launch_bounds__(32)
void oproj_kernel(const _Float16* __restrict__ Ctx,  // [S][HID] f16
                  const _Float16* __restrict__ Woh,  // [HID][HID] f16
                  float* __restrict__ out) {         // [S][HID] f32
    const int lane = threadIdx.x;
    const int r = lane & 15, hf = lane >> 4;
    const int s0 = blockIdx.x * 32;
    const int o0 = blockIdx.y * 64;

    v8f acc[2][4] = {};
    const _Float16* xb = Ctx + (size_t)s0 * HIDN;
    const _Float16* wb = Woh + (size_t)o0 * HIDN;

    for (int k = 0; k < HIDN; k += 32) {
        v16h a0 = load_a(xb + k,                     HIDN, lane);
        v16h a1 = load_a(xb + (size_t)16 * HIDN + k, HIDN, lane);
#pragma unroll
        for (int n = 0; n < 4; ++n) {
            v16h b = load_b(wb + (size_t)n * 16 * HIDN + k, HIDN, lane);
            acc[0][n] = wmma_f16(a0, b, acc[0][n]);
            acc[1][n] = wmma_f16(a1, b, acc[1][n]);
        }
    }
#pragma unroll
    for (int g = 0; g < 2; ++g)
#pragma unroll
        for (int e = 0; e < 8; ++e) {
            const int srow = s0 + g * 16 + e + 8 * hf;
#pragma unroll
            for (int n = 0; n < 4; ++n)
                out[(size_t)srow * HIDN + o0 + n * 16 + r] = acc[g][n][e];
        }
}

// ---------------------------------------------------------------------------
extern "C" void kernel_launch(void* const* d_in, const int* in_sizes, int n_in,
                              void* d_out, int out_size, void* d_ws, size_t ws_size,
                              hipStream_t stream) {
    const float* hs   = (const float*)d_in[0];  // [1,S,HID]
    const float* cosb = (const float*)d_in[1];  // [1,1,S,DH]
    const float* sinb = (const float*)d_in[2];  // [1,1,S,DH]
    const float* Wq   = (const float*)d_in[3];  // [NH*DH, HID]
    const float* Wk   = (const float*)d_in[4];  // [NKV*DH, HID]
    const float* Wv   = (const float*)d_in[5];  // [NKV*DH, HID]
    const float* Wo   = (const float*)d_in[6];  // [HID, NH*DH]
    float* out = (float*)d_out;

    // Workspace carve-up (~48 MB).
    char* ws = (char*)d_ws;
    size_t off = 0;
    auto carve = [&](size_t bytes) -> void* {
        void* p = ws + off;
        off += (bytes + 255) & ~(size_t)255;
        return p;
    };
    _Float16* Xh  = (_Float16*)carve((size_t)S_LEN * HIDN * 2);
    _Float16* Wqh = (_Float16*)carve((size_t)NH  * DH * HIDN * 2);
    _Float16* Wkh = (_Float16*)carve((size_t)NKV * DH * HIDN * 2);
    _Float16* Wvh = (_Float16*)carve((size_t)NKV * DH * HIDN * 2);
    _Float16* Woh = (_Float16*)carve((size_t)HIDN * HIDN * 2);
    _Float16* Qh  = (_Float16*)carve((size_t)NH  * S_LEN * DH * 2);
    _Float16* Kh  = (_Float16*)carve((size_t)NKV * S_LEN * DH * 2);
    _Float16* Vt  = (_Float16*)carve((size_t)NKV * DH * S_LEN * 2);
    _Float16* Ctx = (_Float16*)carve((size_t)S_LEN * HIDN * 2);

    auto cvt = [&](const float* src, _Float16* dst, int n) {
        f32_to_f16_kernel<<<(n + 255) / 256, 256, 0, stream>>>(src, dst, n);
    };
    cvt(hs, Xh,  S_LEN * HIDN);
    cvt(Wq, Wqh, NH  * DH * HIDN);
    cvt(Wk, Wkh, NKV * DH * HIDN);
    cvt(Wv, Wvh, NKV * DH * HIDN);
    cvt(Wo, Woh, HIDN * HIDN);

    // Q / K projections (+RoPE), V projection (transposed store).
    proj_kernel<<<dim3(S_LEN / 32, NH),  32, 0, stream>>>(Xh, Wqh, cosb, sinb, Qh, 0);
    proj_kernel<<<dim3(S_LEN / 32, NKV), 32, 0, stream>>>(Xh, Wkh, cosb, sinb, Kh, 0);
    proj_kernel<<<dim3(S_LEN / 32, NKV), 32, 0, stream>>>(Xh, Wvh, nullptr, nullptr, Vt, 1);

    // Causal GQA attention (32 query rows per wave).
    attn_kernel<<<dim3(S_LEN / 32, NH), 32, 0, stream>>>(Qh, Kh, Vt, Ctx);

    // Output projection.
    oproj_kernel<<<dim3(S_LEN / 32, HIDN / 64), 32, 0, stream>>>(Ctx, Woh, out);
}
